// ImgtoClass_Metric_61435212201998
// MI455X (gfx1250) — matrix operand order repaired and checked by reading, not compiled, not run on hardware
//
#include <hip/hip_runtime.h>
#include <math.h>

// Problem constants (from reference setup_inputs)
#define BB     16          // batch
#define CC     64          // channels (K dim)
#define HW     441         // h*w
#define QPAD   448         // padded query rows
#define QT     28          // query tiles of 16
#define NC     10          // classes
#define MM     2205        // support descriptors per class
#define MPAD   2208
#define MT     138         // m tiles of 16
#define KTILES 2           // 64 = 2 x K32
#define TILEH  (KTILES * 512)   // halves per packed 16-col tile (K=64)
#define NEGINF (-1e30f)

// Packed panel sizes (halves)
#define A_HALVES ((size_t)BB * QT * KTILES * 32 * 16)   // 458752
#define A_BYTES  (A_HALVES * 2)                          // 917504 (256B aligned)

typedef _Float16 v16h __attribute__((ext_vector_type(16)));
typedef float    v8f  __attribute__((ext_vector_type(8)));

// Branchless top-3 insert (sorting network step): keeps t0 >= t1 >= t2.
__device__ __forceinline__ void top3_insert(float& t0, float& t1, float& t2, float val) {
    float mx0 = fmaxf(t0, val);
    float mn0 = fminf(t0, val);
    float mx1 = fmaxf(t1, mn0);
    float mn1 = fminf(t1, mn0);
    t0 = mx0;
    t1 = mx1;
    t2 = fmaxf(t2, mn1);
}

// ---------------------------------------------------------------------------
// Pack + normalize queries into WMMA A-matrix operand order (same k order as pack_s,
// 16-bit A 16x32): lane = g*16 + row; elem i -> K = g*16 + i
// (per-lane-group contiguous).
// ---------------------------------------------------------------------------
__global__ void pack_q(const float* __restrict__ x1, _Float16* __restrict__ Apack) {
    int idx = blockIdx.x * blockDim.x + threadIdx.x;
    if (idx >= BB * QPAD) return;
    int b  = idx / QPAD;
    int hw = idx % QPAD;

    float v[CC];
    float ss = 0.f;
    if (hw < HW) {
        const float* p = x1 + (size_t)b * CC * HW + hw;
#pragma unroll
        for (int c = 0; c < CC; ++c) { float t = p[(size_t)c * HW]; v[c] = t; ss += t * t; }
    } else {
#pragma unroll
        for (int c = 0; c < CC; ++c) v[c] = 0.f;
    }
    float inv = (hw < HW) ? (1.f / fmaxf(sqrtf(ss), 1e-12f)) : 0.f;

    int qt = hw >> 4, r = hw & 15;
#pragma unroll
    for (int k = 0; k < CC; ++k) {
        int kt = k >> 5, kk = k & 31;
        int g, e;
        g = kk >> 4; e = kk & 15;
        int lane = g * 16 + r;
        size_t off = ((((size_t)b * QT + qt) * KTILES + kt) * 32 + lane) * 16 + e;
        Apack[off] = (_Float16)(v[k] * inv);
    }
}

// ---------------------------------------------------------------------------
// Pack + normalize supports into WMMA B-matrix register order (16-bit B 32x16):
// lane = g*16 + col; elem i -> K = g*16 + i (per-lane-group contiguous).
// ---------------------------------------------------------------------------
__global__ void pack_s(const float* __restrict__ x2, _Float16* __restrict__ Bpack) {
    int idx = blockIdx.x * blockDim.x + threadIdx.x;
    if (idx >= NC * MPAD) return;
    int n = idx / MPAD;
    int m = idx % MPAD;

    float v[CC];
    float ss = 0.f;
    if (m < MM) {
        const float* p = x2 + (size_t)n * CC * MM + m;
#pragma unroll
        for (int c = 0; c < CC; ++c) { float t = p[(size_t)c * MM]; v[c] = t; ss += t * t; }
    } else {
#pragma unroll
        for (int c = 0; c < CC; ++c) v[c] = 0.f;
    }
    float inv = (m < MM) ? (1.f / fmaxf(sqrtf(ss), 1e-12f)) : 0.f;

    int mt = m >> 4, col = m & 15;
#pragma unroll
    for (int k = 0; k < CC; ++k) {
        int kt = k >> 5, kk = k & 31;
        int g = kk >> 4, e = kk & 15;
        int lane = g * 16 + col;
        size_t off = ((((size_t)n * MT + mt) * KTILES + kt) * 32 + lane) * 16 + e;
        Bpack[off] = (_Float16)(v[k] * inv);
    }
}

__global__ void zero_out(float* __restrict__ out, int n) {
    int i = blockIdx.x * blockDim.x + threadIdx.x;
    if (i < n) out[i] = 0.f;
}

// ---------------------------------------------------------------------------
// Fused GEMM + per-row top-3 + sum. One wave per 16-row q tile; block of 4
// waves covers 4 consecutive q tiles of one (b, n) pair; loop over 138 m
// tiles with K=64 via two chained v_wmma_f32_16x16x32_f16.
// C/D layout: lane l, VGPR v holds D[8*(l>>4)+v][l&15].
// Hot loop: unrolled x2 with two independent B buffer sets (no loop-carried
// register copies); each buffer is consumed then immediately reloaded two
// tiles ahead, so s_wait_loadcnt lands a full tile of WMMA+VALU later.
// ---------------------------------------------------------------------------
__global__ __launch_bounds__(128) void gemm_topk(const _Float16* __restrict__ Apack,
                                                 const _Float16* __restrict__ Bpack,
                                                 float* __restrict__ out) {
    const int wave = threadIdx.x >> 5;
    const int lane = threadIdx.x & 31;
    const int col  = lane & 15;
    const int g    = lane >> 4;

    int blk = blockIdx.x;                 // [b][n][qgroup]
    int qg  = blk % 7;  blk /= 7;
    int n   = blk % NC;
    int b   = blk / NC;
    int qt  = qg * 4 + wave;              // 0..27

    // A panel for this q tile: resident in registers for the whole m loop.
    const _Float16* Abase = Apack + (((size_t)b * QT + qt) * KTILES) * 512 + (size_t)lane * 16;
    v16h a0 = *(const v16h*)(Abase);
    v16h a1 = *(const v16h*)(Abase + 512);

    float t0[8], t1[8], t2[8];
#pragma unroll
    for (int v = 0; v < 8; ++v) { t0[v] = NEGINF; t1[v] = NEGINF; t2[v] = NEGINF; }

    const _Float16* Bn = Bpack + ((size_t)n * MT) * TILEH + (size_t)lane * 16;

    // Preload tiles 0 (p) and 1 (q).
    v16h p0 = *(const v16h*)(Bn);
    v16h p1 = *(const v16h*)(Bn + 512);
    v16h q0 = *(const v16h*)(Bn + TILEH);
    v16h q1 = *(const v16h*)(Bn + TILEH + 512);

    // Pair loop over full tiles (0,1)...(134,135); loads reach tiles <= 137,
    // all inside the padded Bpack, so no guards in the hot loop.
    for (int mt = 0; mt < MT - 2; mt += 2) {
        const _Float16* Bp2 = Bn + ((size_t)mt + 2) * TILEH;
        __builtin_prefetch(Bp2 + 2 * TILEH, 0, 1);   // 4 tiles ahead (speculative)

        // Tile mt from buffer p; reload p with tile mt+2.
        v8f c = {};
        c = __builtin_amdgcn_wmma_f32_16x16x32_f16(false, a0, false, p0,
                                                   (short)0, c, false, false);
        c = __builtin_amdgcn_wmma_f32_16x16x32_f16(false, a1, false, p1,
                                                   (short)0, c, false, false);
        p0 = *(const v16h*)(Bp2);
        p1 = *(const v16h*)(Bp2 + 512);
#pragma unroll
        for (int v = 0; v < 8; ++v)
            top3_insert(t0[v], t1[v], t2[v], c[v]);

        // Tile mt+1 from buffer q; reload q with tile mt+3.
        v8f d = {};
        d = __builtin_amdgcn_wmma_f32_16x16x32_f16(false, a0, false, q0,
                                                   (short)0, d, false, false);
        d = __builtin_amdgcn_wmma_f32_16x16x32_f16(false, a1, false, q1,
                                                   (short)0, d, false, false);
        q0 = *(const v16h*)(Bp2 + TILEH);
        q1 = *(const v16h*)(Bp2 + TILEH + 512);
#pragma unroll
        for (int v = 0; v < 8; ++v)
            top3_insert(t0[v], t1[v], t2[v], d[v]);
    }

    // Tile 136 (full) from buffer p.
    {
        v8f c = {};
        c = __builtin_amdgcn_wmma_f32_16x16x32_f16(false, a0, false, p0,
                                                   (short)0, c, false, false);
        c = __builtin_amdgcn_wmma_f32_16x16x32_f16(false, a1, false, p1,
                                                   (short)0, c, false, false);
#pragma unroll
        for (int v = 0; v < 8; ++v)
            top3_insert(t0[v], t1[v], t2[v], c[v]);
    }

    // Tile 137 (peeled) from buffer q: columns 13..15 are padding -> -inf.
    {
        v8f c = {};
        c = __builtin_amdgcn_wmma_f32_16x16x32_f16(false, a0, false, q0,
                                                   (short)0, c, false, false);
        c = __builtin_amdgcn_wmma_f32_16x16x32_f16(false, a1, false, q1,
                                                   (short)0, c, false, false);
        const bool valid = ((MT - 1) * 16 + col) < MM;
#pragma unroll
        for (int v = 0; v < 8; ++v) {
            float val = valid ? c[v] : NEGINF;
            top3_insert(t0[v], t1[v], t2[v], val);
        }
    }

    // Merge per-lane top-3 lists into per-row global top-3 via LDS.
    __shared__ float red[4][16][16][3];
    __shared__ float rowsum[4][16];
#pragma unroll
    for (int v = 0; v < 8; ++v) {
        int row = g * 8 + v;
        red[wave][row][col][0] = t0[v];
        red[wave][row][col][1] = t1[v];
        red[wave][row][col][2] = t2[v];
    }
    __syncthreads();

    if (lane < 16) {
        float m0 = NEGINF, m1 = NEGINF, m2 = NEGINF;
        for (int cc = 0; cc < 16; ++cc) {
#pragma unroll
            for (int j = 0; j < 3; ++j)
                top3_insert(m0, m1, m2, red[wave][lane][cc][j]);
        }
        rowsum[wave][lane] = m0 + m1 + m2;   // padded q rows contribute 0
    }
    __syncthreads();

    if (lane == 0) {
        float s = 0.f;
#pragma unroll
        for (int r = 0; r < 16; ++r) s += rowsum[wave][r];
        atomicAdd(&out[b * NC + n], s);
    }
}

extern "C" void kernel_launch(void* const* d_in, const int* in_sizes, int n_in,
                              void* d_out, int out_size, void* d_ws, size_t ws_size,
                              hipStream_t stream) {
    const float* x1 = (const float*)d_in[0];   // [16,64,21,21] f32
    const float* x2 = (const float*)d_in[1];   // [10,64,2205] f32
    float* out = (float*)d_out;                // [16,10] f32

    _Float16* Apack = (_Float16*)d_ws;
    _Float16* Bpack = (_Float16*)((char*)d_ws + A_BYTES);

    pack_q<<<(BB * QPAD + 255) / 256, 256, 0, stream>>>(x1, Apack);
    pack_s<<<(NC * MPAD + 255) / 256, 256, 0, stream>>>(x2, Bpack);
    zero_out<<<1, 256, 0, stream>>>(out, BB * NC);
    gemm_topk<<<BB * NC * 7, 128, 0, stream>>>(Apack, Bpack, out);
}
